// Model_10333691314799
// MI455X (gfx1250) — compile-verified, matrix operand order checked by reading, not run on hardware
//
#include <hip/hip_runtime.h>

#define NN 100000
#define EE 1600000
#define VV 50000
#define HH 256
#define AA 64
#define NHD 3
#define GG 64

typedef __attribute__((ext_vector_type(16))) _Float16 v16h;
typedef __attribute__((ext_vector_type(4)))  _Float16 v4h;
typedef __attribute__((ext_vector_type(8)))  float    v8f;

union FragU { v16h v; unsigned u[8]; };

// K offset inside a 32-wide K tile for 16-bit A/B WMMA fragments (ISA 7.12.2):
// VGPR i (i<4): K = half*8 + 2*(i&3); VGPR i (i>=4): K = 16 + half*8 + 2*(i&3)
__device__ __forceinline__ int kbase(int i, int h) {
  return ((i & 4) << 2) + (h << 3) + ((i & 3) << 1);
}

__device__ __forceinline__ void atomicMaxF(float* addr, float val) {
  if (val >= 0.f) atomicMax((int*)addr, __float_as_int(val));
  else            atomicMin((unsigned int*)addr, __float_as_uint(val));
}

// ---------------- prep kernels ----------------

__global__ void f32_to_f16_kernel(const float* __restrict__ in,
                                  _Float16* __restrict__ out, int n) {
  int i = blockIdx.x * blockDim.x + threadIdx.x;  // n multiple of 4
  if (i * 4 >= n) return;
  const float4 f = ((const float4*)in)[i];
  v4h o; o.x = (_Float16)f.x; o.y = (_Float16)f.y; o.z = (_Float16)f.z; o.w = (_Float16)f.w;
  ((v4h*)out)[i] = o;
}

// Pack a KxNcols f32 weight matrix into WMMA B-fragment layout:
// frag[((nt*nKT + kt)*32 + lane)*8 + i] = pack_f16( W[k][col], W[k+1][col] )
// col = nt*16 + (lane&15), k = kt*32 + kbase(i, lane>>4)
__global__ void pack_b_frags(const float* __restrict__ W,
                             unsigned* __restrict__ frag,
                             int Ncols, int nKT, int total) {
  int idx = blockIdx.x * blockDim.x + threadIdx.x;
  if (idx >= total) return;
  int i    = idx & 7;
  int lane = (idx >> 3) & 31;
  int rem  = idx >> 8;
  int kt   = rem % nKT;
  int nt   = rem / nKT;
  int col  = nt * 16 + (lane & 15);
  int h    = lane >> 4;
  int k    = kt * 32 + kbase(i, h);
  _Float16 lo = (_Float16)W[k * Ncols + col];
  _Float16 hi = (_Float16)W[(k + 1) * Ncols + col];
  unsigned short us0, us1;
  __builtin_memcpy(&us0, &lo, 2);
  __builtin_memcpy(&us1, &hi, 2);
  frag[idx] = (unsigned)us0 | ((unsigned)us1 << 16);
}

// ---------------- GAT projection: feat = emb[h0] @ W_fc  (WMMA) ----------------
// One wave per block; 16 nodes per block; K = 256 (8 k-steps), 12 n-tiles.
__global__ void __launch_bounds__(32) gemm_feat(const int* __restrict__ h0,
                                                const _Float16* __restrict__ emb_h,
                                                const unsigned* __restrict__ wfc_frag,
                                                float* __restrict__ feat) {
  __shared__ _Float16 sA[16 * HH];
  const int lane = threadIdx.x;
  const int base = blockIdx.x * 16;
  // vectorized gather: 16 rows x 32 chunks of 16B; each iteration = 1 full row coalesced
  {
    uint4* s4 = (uint4*)sA;
#pragma unroll
    for (int it = 0; it < 16; ++it) {
      int c = it * 32 + lane;         // chunk id 0..511
      int r = c >> 5;                 // 32 chunks per row
      int k8 = c & 31;
      const uint4* row = (const uint4*)(emb_h + (long long)h0[base + r] * HH);
      s4[c] = row[k8];
    }
  }
  __syncthreads();
  const int row = lane & 15, hh = lane >> 4;
  for (int nt = 0; nt < 12; ++nt) {
    v8f acc = {};
    for (int kt = 0; kt < 8; ++kt) {
      FragU a, b;
      const unsigned* bp = wfc_frag + (((nt * 8) + kt) * 32 + lane) * 8;
#pragma unroll
      for (int i = 0; i < 8; ++i) {
        a.u[i] = *(const unsigned*)&sA[row * HH + kt * 32 + kbase(i, hh)];
        b.u[i] = bp[i];
      }
      acc = __builtin_amdgcn_wmma_f32_16x16x32_f16(
          false, a.v, false, b.v, (short)0, acc, false, false);
    }
#pragma unroll
    for (int j = 0; j < 8; ++j)  // D: (M = j + 8*hh, Ncol = nt*16 + row)
      feat[(long long)(base + j + 8 * hh) * 192 + nt * 16 + row] = acc[j];
  }
}

// ---------------- attention scalars + softmax state init ----------------

__global__ void compute_elr(const float* __restrict__ feat,
                            const float* __restrict__ attn_l,
                            const float* __restrict__ attn_r,
                            float* __restrict__ el, float* __restrict__ er,
                            float* __restrict__ mbuf, float* __restrict__ denom) {
  int idx = blockIdx.x * blockDim.x + threadIdx.x;   // n*3 + h
  if (idx >= NN * NHD) return;
  int h = idx % NHD, n = idx / NHD;
  const float4* f  = (const float4*)(feat + (long long)n * 192 + h * AA);
  const float4* al = (const float4*)(attn_l + h * AA);
  const float4* ar = (const float4*)(attn_r + h * AA);
  float sl = 0.f, sr = 0.f;
#pragma unroll
  for (int a = 0; a < AA / 4; ++a) {
    float4 fv = f[a], lv = al[a], rv = ar[a];
    sl += fv.x * lv.x + fv.y * lv.y + fv.z * lv.z + fv.w * lv.w;
    sr += fv.x * rv.x + fv.y * rv.y + fv.z * rv.z + fv.w * rv.w;
  }
  el[idx] = sl; er[idx] = sr;
  mbuf[idx] = -__builtin_inff();
  denom[idx] = 0.f;
}

__global__ void edge_max(const int* __restrict__ src, const int* __restrict__ dst,
                         const float* __restrict__ el, const float* __restrict__ er,
                         float* __restrict__ mbuf) {
  int idx = blockIdx.x * blockDim.x + threadIdx.x;   // e*3 + h
  if (idx >= EE * NHD) return;
  int h = idx % NHD, e = idx / NHD;
  float v = el[src[e] * NHD + h] + er[dst[e] * NHD + h];
  v = v > 0.f ? v : 0.2f * v;
  atomicMaxF(&mbuf[dst[e] * NHD + h], v);
}

__global__ void edge_expsum(const int* __restrict__ src, const int* __restrict__ dst,
                            const float* __restrict__ el, const float* __restrict__ er,
                            const float* __restrict__ mbuf, float* __restrict__ denom) {
  int idx = blockIdx.x * blockDim.x + threadIdx.x;
  if (idx >= EE * NHD) return;
  int h = idx % NHD, e = idx / NHD;
  int d = dst[e];
  float v = el[src[e] * NHD + h] + er[d * NHD + h];
  v = v > 0.f ? v : 0.2f * v;
  atomicAdd(&denom[d * NHD + h], __expf(v - mbuf[d * NHD + h]));
}

__global__ void edge_alpha(const int* __restrict__ src, const int* __restrict__ dst,
                           const float* __restrict__ el, const float* __restrict__ er,
                           const float* __restrict__ mbuf, const float* __restrict__ denom,
                           float* __restrict__ alpha) {
  int idx = blockIdx.x * blockDim.x + threadIdx.x;
  if (idx >= EE * NHD) return;
  int h = idx % NHD, e = idx / NHD;
  int d = dst[e];
  float v = el[src[e] * NHD + h] + er[d * NHD + h];
  v = v > 0.f ? v : 0.2f * v;
  alpha[idx] = __expf(v - mbuf[d * NHD + h]) / denom[d * NHD + h];
}

// rst[dst] += alpha * feat[src]; one thread per float4 chunk (48 chunks/edge).
// Roofline limiter: ~2.5 GB of feat reads + atomic adds.
__global__ void edge_scatter(const int* __restrict__ src, const int* __restrict__ dst,
                             const float* __restrict__ alpha,
                             const float* __restrict__ feat, float* __restrict__ rst) {
  long long idx = (long long)blockIdx.x * blockDim.x + threadIdx.x; // e*48 + c4
  if (idx >= (long long)EE * 48) return;
  int c4 = (int)(idx % 48);
  int e  = (int)(idx / 48);
  int c  = c4 * 4;               // 0..188, head boundaries (64) are multiples of 4
  int h  = c >> 6;
  float a = alpha[e * NHD + h];
  const float4 f = *(const float4*)&feat[(long long)src[e] * 192 + c];
  float* r = &rst[(long long)dst[e] * 192 + c];
  atomicAdd(r + 0, a * f.x);
  atomicAdd(r + 1, a * f.y);
  atomicAdd(r + 2, a * f.z);
  atomicAdd(r + 3, a * f.w);
}

// bias + relu + mean over heads -> h2 (stored f16 for MLP WMMA); float4 per thread
__global__ void head_mean(const float* __restrict__ rst, const float* __restrict__ bias,
                          _Float16* __restrict__ h2h) {
  int idx = blockIdx.x * blockDim.x + threadIdx.x;  // n*16 + a4
  if (idx >= NN * 16) return;
  int a4 = idx & 15, n = idx >> 4;
  int a = a4 * 4;
  float sx = 0.f, sy = 0.f, sz = 0.f, sw = 0.f;
#pragma unroll
  for (int h = 0; h < NHD; ++h) {
    float4 rv = *(const float4*)&rst[(long long)n * 192 + h * AA + a];
    float4 bv = *(const float4*)&bias[h * AA + a];
    float vx = rv.x + bv.x, vy = rv.y + bv.y, vz = rv.z + bv.z, vw = rv.w + bv.w;
    sx += vx > 0.f ? vx : 0.f;
    sy += vy > 0.f ? vy : 0.f;
    sz += vz > 0.f ? vz : 0.f;
    sw += vw > 0.f ? vw : 0.f;
  }
  const float k = 1.f / 3.f;
  v4h o; o.x = (_Float16)(sx * k); o.y = (_Float16)(sy * k);
  o.z = (_Float16)(sz * k); o.w = (_Float16)(sw * k);
  *(v4h*)&h2h[(long long)n * AA + a] = o;
}

// ---------------- fused MLP: relu(h2@W1+b1)@W2+b2  (WMMA x2) ----------------

__global__ void __launch_bounds__(32) mlp_wmma(const _Float16* __restrict__ h2h,
                                               const unsigned* __restrict__ w1f,
                                               const unsigned* __restrict__ w2f,
                                               const float* __restrict__ b1,
                                               const float* __restrict__ b2,
                                               float* __restrict__ h3) {
  __shared__ _Float16 sH[16 * 64];
  __shared__ _Float16 sT[16 * 128];
  const int lane = threadIdx.x;
  const int base = blockIdx.x * 16;
  {
    uint4* s4 = (uint4*)sH;
    const uint4* g4 = (const uint4*)(h2h + (long long)base * 64);
#pragma unroll
    for (int it = 0; it < 4; ++it) s4[it * 32 + lane] = g4[it * 32 + lane];
  }
  __syncthreads();
  const int row = lane & 15, hh = lane >> 4;
  // GEMM1: (16x64)@(64x128), K = 2 steps, 8 n-tiles; +b1, relu -> LDS f16
  for (int nt = 0; nt < 8; ++nt) {
    v8f acc = {};
#pragma unroll
    for (int kt = 0; kt < 2; ++kt) {
      FragU a, b;
      const unsigned* bp = w1f + ((nt * 2 + kt) * 32 + lane) * 8;
#pragma unroll
      for (int i = 0; i < 8; ++i) {
        a.u[i] = *(const unsigned*)&sH[row * 64 + kt * 32 + kbase(i, hh)];
        b.u[i] = bp[i];
      }
      acc = __builtin_amdgcn_wmma_f32_16x16x32_f16(
          false, a.v, false, b.v, (short)0, acc, false, false);
    }
    int col = nt * 16 + row;
    float bv = b1[col];
#pragma unroll
    for (int j = 0; j < 8; ++j) {
      float v = acc[j] + bv;
      sT[(j + 8 * hh) * 128 + col] = (_Float16)(v > 0.f ? v : 0.f);
    }
  }
  __syncthreads();
  // GEMM2: (16x128)@(128x16), K = 4 steps, 1 n-tile; +b2
  v8f acc = {};
#pragma unroll
  for (int kt = 0; kt < 4; ++kt) {
    FragU a, b;
    const unsigned* bp = w2f + (kt * 32 + lane) * 8;
#pragma unroll
    for (int i = 0; i < 8; ++i) {
      a.u[i] = *(const unsigned*)&sT[row * 128 + kt * 32 + kbase(i, hh)];
      b.u[i] = bp[i];
    }
    acc = __builtin_amdgcn_wmma_f32_16x16x32_f16(
        false, a.v, false, b.v, (short)0, acc, false, false);
  }
  float bv = b2[row];
#pragma unroll
  for (int j = 0; j < 8; ++j)
    h3[(long long)(base + j + 8 * hh) * 16 + row] = acc[j] + bv;
}

// ---------------- per-graph mean readout ----------------

__global__ void readout_acc(const float* __restrict__ h3, const int* __restrict__ gid,
                            float* __restrict__ sums, float* __restrict__ counts) {
  int n = blockIdx.x * blockDim.x + threadIdx.x;
  if (n >= NN) return;
  int g = gid[n];
  atomicAdd(&counts[g], 1.f);
  const float4* hv = (const float4*)(h3 + (long long)n * 16);
#pragma unroll
  for (int q = 0; q < 4; ++q) {
    float4 v = hv[q];
    atomicAdd(&sums[g * 16 + q * 4 + 0], v.x);
    atomicAdd(&sums[g * 16 + q * 4 + 1], v.y);
    atomicAdd(&sums[g * 16 + q * 4 + 2], v.z);
    atomicAdd(&sums[g * 16 + q * 4 + 3], v.w);
  }
}

__global__ void readout_div(const float* __restrict__ sums, const float* __restrict__ counts,
                            float* __restrict__ out) {
  int idx = threadIdx.x;
  if (idx >= GG * 16) return;
  out[idx] = sums[idx] / fmaxf(counts[idx >> 4], 1.f);
}

// ---------------- launcher ----------------

extern "C" void kernel_launch(void* const* d_in, const int* in_sizes, int n_in,
                              void* d_out, int out_size, void* d_ws, size_t ws_size,
                              hipStream_t stream) {
  (void)in_sizes; (void)n_in; (void)out_size; (void)ws_size;
  const int*   h0     = (const int*)d_in[0];
  const int*   src    = (const int*)d_in[1];
  const int*   dst    = (const int*)d_in[2];
  const int*   gid    = (const int*)d_in[3];
  const float* emb    = (const float*)d_in[4];
  const float* Wfc    = (const float*)d_in[5];
  const float* attn_l = (const float*)d_in[6];
  const float* attn_r = (const float*)d_in[7];
  const float* bias   = (const float*)d_in[8];
  const float* W1     = (const float*)d_in[9];
  const float* b1     = (const float*)d_in[10];
  const float* W2     = (const float*)d_in[11];
  const float* b2     = (const float*)d_in[12];
  float* out = (float*)d_out;

  char* ws = (char*)d_ws;
  size_t off = 0;
  auto alloc = [&](size_t bytes) -> void* {
    void* p = ws + off;
    off += (bytes + 255) & ~(size_t)255;
    return p;
  };
  _Float16* emb_h   = (_Float16*)alloc((size_t)VV * HH * 2);
  unsigned* wfc_f   = (unsigned*)alloc((size_t)12 * 8 * 32 * 8 * 4);
  unsigned* w1_f    = (unsigned*)alloc((size_t)8 * 2 * 32 * 8 * 4);
  unsigned* w2_f    = (unsigned*)alloc((size_t)1 * 4 * 32 * 8 * 4);
  float*    feat    = (float*)alloc((size_t)NN * 192 * 4);
  float*    el      = (float*)alloc((size_t)NN * NHD * 4);
  float*    er      = (float*)alloc((size_t)NN * NHD * 4);
  float*    mbuf    = (float*)alloc((size_t)NN * NHD * 4);
  float*    denom   = (float*)alloc((size_t)NN * NHD * 4);
  float*    alpha   = (float*)alloc((size_t)EE * NHD * 4);
  float*    rst     = (float*)alloc((size_t)NN * 192 * 4);
  _Float16* h2h     = (_Float16*)alloc((size_t)NN * AA * 2);
  float*    h3      = (float*)alloc((size_t)NN * 16 * 4);
  float*    sums    = (float*)alloc((size_t)GG * 16 * 4);
  float*    counts  = (float*)alloc((size_t)GG * 4);

  // zero accumulators (graph-capture safe)
  hipMemsetAsync(rst, 0, (size_t)NN * 192 * 4, stream);
  hipMemsetAsync(sums, 0, (size_t)GG * 16 * 4, stream);
  hipMemsetAsync(counts, 0, (size_t)GG * 4, stream);

  // prep: emb->f16 (vectorized, VV*HH divisible by 4), weight fragment packing
  {
    int n = VV * HH;
    f32_to_f16_kernel<<<(n / 4 + 255) / 256, 256, 0, stream>>>(emb, emb_h, n);
  }
  pack_b_frags<<<(12 * 8 * 256 + 255) / 256, 256, 0, stream>>>(Wfc, wfc_f, 192, 8, 12 * 8 * 256);
  pack_b_frags<<<(8 * 2 * 256 + 255) / 256, 256, 0, stream>>>(W1, w1_f, 128, 2, 8 * 2 * 256);
  pack_b_frags<<<(1 * 4 * 256 + 255) / 256, 256, 0, stream>>>(W2, w2_f, 16, 4, 1 * 4 * 256);

  // GAT projection (WMMA)
  gemm_feat<<<NN / 16, 32, 0, stream>>>(h0, emb_h, wfc_f, feat);

  // attention
  compute_elr<<<(NN * NHD + 255) / 256, 256, 0, stream>>>(feat, attn_l, attn_r, el, er, mbuf, denom);
  edge_max<<<(EE * NHD + 255) / 256, 256, 0, stream>>>(src, dst, el, er, mbuf);
  edge_expsum<<<(EE * NHD + 255) / 256, 256, 0, stream>>>(src, dst, el, er, mbuf, denom);
  edge_alpha<<<(EE * NHD + 255) / 256, 256, 0, stream>>>(src, dst, el, er, mbuf, denom, alpha);
  {
    long long total = (long long)EE * 48;
    int blocks = (int)((total + 255) / 256);
    edge_scatter<<<blocks, 256, 0, stream>>>(src, dst, alpha, feat, rst);
  }
  head_mean<<<(NN * 16 + 255) / 256, 256, 0, stream>>>(rst, bias, h2h);

  // MLP (WMMA x2, fused)
  mlp_wmma<<<NN / 16, 32, 0, stream>>>(h2h, w1_f, w2_f, b1, b2, h3);

  // per-graph mean
  readout_acc<<<(NN + 255) / 256, 256, 0, stream>>>(h3, gid, sums, counts);
  readout_div<<<1, 1024, 0, stream>>>(sums, counts, out);
}
